// CausalGraphVAE_34084860461442
// MI455X (gfx1250) — matrix operand order repaired and checked by reading, not compiled, not run on hardware
//
#include <hip/hip_runtime.h>
#include <hip/hip_bf16.h>
#include <stdint.h>
#include <stddef.h>

// ---------------------------------------------------------------------------
// CausalGraphVAE forward for MI455X (gfx1250)
//   - v_wmma_f32_16x16x32_bf16 for all GEMMs (fp32 accumulate)
//   - B panels staged to LDS via async-to-LDS DMA (double buffered, ASYNCcnt)
//   - each wave: 2 M-tiles x 4 N-tiles = 8 WMMAs per 32-K step
// ---------------------------------------------------------------------------

typedef __bf16 bf16_t;
typedef bf16_t bf16x16 __attribute__((ext_vector_type(16)));
typedef bf16_t bf16x8  __attribute__((ext_vector_type(8)));
typedef float  f32x8   __attribute__((ext_vector_type(8)));
typedef int    i32x4_vs __attribute__((vector_size(16)));   // builtin's pointee type

#define NN   4096   // num nodes
#define DIN  256
#define HIDD 256
#define LATD 64
#define DH   768    // D_IN + 2*HID

// --- CDNA5 async-to-LDS path (guarded: falls back to plain LDS staging) ----
#if __has_builtin(__builtin_amdgcn_global_load_async_to_lds_b128)
#define USE_ASYNC_STAGE 1
#define ASYNC_B128(gsrc, ldst)                                              \
    __builtin_amdgcn_global_load_async_to_lds_b128(                         \
        (__attribute__((address_space(1))) i32x4_vs*)(gsrc),                \
        (__attribute__((address_space(3))) i32x4_vs*)(ldst), 0, 0)
#else
#define USE_ASYNC_STAGE 0
#endif

#if __has_builtin(__builtin_amdgcn_s_wait_asynccnt)
#define WAIT_ASYNC(n) __builtin_amdgcn_s_wait_asynccnt(n)
#else
#define WAIT_ASYNC(n) asm volatile("s_wait_asynccnt %0" ::"n"(n) : "memory")
#endif

// ---------------------------------------------------------------------------
// Elementwise / prep kernels
// ---------------------------------------------------------------------------

__device__ __forceinline__ float sigf(float x) { return 1.0f / (1.0f + __expf(-x)); }

// deg[j] = sum_i sigmoid(es[i][j]); dinv[j] = rsqrt(deg)
__global__ __launch_bounds__(256) void deg_kernel(const float* __restrict__ es,
                                                  float* __restrict__ dinv) {
    int j = blockIdx.x * blockDim.x + threadIdx.x;   // coalesced over j
    float s = 0.0f;
    for (int i = 0; i < NN; ++i) s += sigf(es[(size_t)i * NN + j]);
    dinv[j] = (s > 0.0f) ? rsqrtf(s) : 0.0f;
}

// adj[i][j] = sigmoid(es[i][j])  (clip to [0,1] is a no-op for sigmoid)
__global__ __launch_bounds__(256) void adj_kernel(const float* __restrict__ es,
                                                  float* __restrict__ adj, int total) {
    int idx = blockIdx.x * blockDim.x + threadIdx.x;
    int stride = gridDim.x * blockDim.x;
    for (; idx < total; idx += stride) adj[idx] = sigf(es[idx]);
}

// anormT[i][j] = dinv[i] * sigmoid(es[j][i]) * dinv[j]   (bf16, row-major)
__global__ __launch_bounds__(256) void anormt_kernel(const float* __restrict__ es,
                                                     const float* __restrict__ dinv,
                                                     bf16_t* __restrict__ anT) {
    __shared__ float tile[32][33];
    const int bi = blockIdx.x * 32;
    const int bj = blockIdx.y * 32;
    const int tx = threadIdx.x & 31;
    const int ty = threadIdx.x >> 5;
#pragma unroll
    for (int p = 0; p < 4; ++p) {
        int row = bj + ty + p * 8;
        tile[ty + p * 8][tx] = sigf(es[(size_t)row * NN + bi + tx]);
    }
    __syncthreads();
#pragma unroll
    for (int p = 0; p < 4; ++p) {
        int i = bi + ty + p * 8;
        int j = bj + tx;
        anT[(size_t)i * NN + j] = (bf16_t)(dinv[i] * dinv[j] * tile[tx][ty + p * 8]);
    }
}

// fp32 row-major [M,K] -> bf16 packed into dst (leading dim dstld, at coloff)
__global__ __launch_bounds__(256) void pack_rows_kernel(const float* __restrict__ src,
                                                        bf16_t* __restrict__ dst,
                                                        int M, int K, int dstld, int coloff) {
    int idx = blockIdx.x * blockDim.x + threadIdx.x;
    if (idx >= M * K) return;
    int m = idx / K, k = idx - m * K;
    dst[(size_t)m * dstld + coloff + k] = (bf16_t)src[idx];
}

// fp32 row-major W[K][Nn] -> bf16 column-major Wc[n*K + k] (top-K rows)
__global__ __launch_bounds__(256) void wtrans_kernel(const float* __restrict__ W,
                                                     bf16_t* __restrict__ Wc,
                                                     int K, int Nn) {
    int idx = blockIdx.x * blockDim.x + threadIdx.x;
    if (idx >= K * Nn) return;
    int n = idx / K, k = idx - n * K;
    Wc[idx] = (bf16_t)W[(size_t)k * Nn + n];
}

// out = relu((1 - sigmoid(zl)) * tanh(hl))
template <bool OUT_BF16>
__global__ __launch_bounds__(256) void gate_kernel(const float* __restrict__ zl,
                                                   const float* __restrict__ hl,
                                                   void* __restrict__ out, int total) {
    int idx = blockIdx.x * blockDim.x + threadIdx.x;
    if (idx >= total) return;
    float z = sigf(zl[idx]);
    float v = fmaxf((1.0f - z) * tanhf(hl[idx]), 0.0f);
    if (OUT_BF16) ((bf16_t*)out)[idx] = (bf16_t)v;
    else          ((float*)out)[idx]  = v;
}

// z = mu + eps * exp(0.5*logvar); deterministic splitmix64 + Box-Muller
__global__ __launch_bounds__(256) void sample_kernel(const float* __restrict__ mu,
                                                     const float* __restrict__ lv,
                                                     bf16_t* __restrict__ z, int total) {
    int idx = blockIdx.x * blockDim.x + threadIdx.x;
    if (idx >= total) return;
    uint64_t s = (uint64_t)idx * 0x9E3779B97F4A7C15ULL + 0x42ULL;
    s ^= s >> 30; s *= 0xBF58476D1CE4E5B9ULL;
    s ^= s >> 27; s *= 0x94D049BB133111EBULL;
    s ^= s >> 31;
    float u1 = ((uint32_t)s + 1u) * 2.3283064365386963e-10f;   // (0,1]
    float u2 = (uint32_t)(s >> 32) * 2.3283064365386963e-10f;
    float eps = sqrtf(-2.0f * __logf(u1)) * __cosf(6.283185307179586f * u2);
    z[idx] = (bf16_t)(mu[idx] + eps * __expf(0.5f * lv[idx]));
}

// ---------------------------------------------------------------------------
// WMMA bf16 GEMM with LDS-staged (async, double-buffered) B panels.
//   C = act(A @ B + bias)
//   A:  bf16 row-major [M,K], leading dim lda  (M multiple of 256)
//   Bc: bf16 column-major (Bc[n*ldb + k]), ldb == K  (Nn multiple of 64)
//   K:  multiple of 128, or a single stage of 64
//   STORE: 0 = fp32 row-major, 1 = bf16 row-major, 2 = bf16 col-major (ldc=M)
// Block = 256 threads = 8 waves. Block tile: 256 M x 64 N.
// Wave tile: 32 M x 64 N  (2 A fragments reused over 4 shared B fragments).
// B panel (64 cols x <=128 K) staged once per block into LDS, stride-136
// padding for conflict-free ds reads; DMA of stage s+1 overlaps WMMAs of s.
// ---------------------------------------------------------------------------
#define KSTAGE 128
#define BSTRIDE 136   // KSTAGE + 8 pad (bank-conflict-free fragment reads)

template <int ACT, int STORE, bool BIAS>
__global__ __launch_bounds__(256) void gemm_wmma_bf16(const bf16_t* __restrict__ A, int lda,
                                                      const bf16_t* __restrict__ Bc, int ldb,
                                                      const float* __restrict__ bias,
                                                      void* __restrict__ Cout, int ldc,
                                                      int M, int Nn, int K) {
    __shared__ bf16_t bstage[2][64 * BSTRIDE];   // 2 x 17 KB

    const int tid  = threadIdx.x;
    const int wave = tid >> 5;
    const int lane = tid & 31;
    const int half = lane >> 4;          // 0 or 1
    const int r    = lane & 15;
    const int m0   = blockIdx.y * 256 + wave * 32;   // wave covers 2 M-tiles
    const int nbase = blockIdx.x * 64;
    if (m0 >= M) return;                 // uniform per wave (never hits here)

    // ---- B panel staging: 16B chunk cid -> (col c, 8-elem group kq) ----
    auto stage = [&](int buf, int k0, int kc) {
        const int cpc = kc >> 3;                       // 16B chunks per column
        const int sh  = (cpc == 16) ? 4 : 3;           // kc = 128 or 64
        const int chunks = 64 << sh;
        for (int cid = tid; cid < chunks; cid += 256) {
            const int c  = cid >> sh;
            const int kq = cid & (cpc - 1);
            const bf16_t* gsrc = Bc + (size_t)(nbase + c) * ldb + k0 + kq * 8;
            bf16_t* ldst = &bstage[buf][c * BSTRIDE + kq * 8];
#if USE_ASYNC_STAGE
            ASYNC_B128(gsrc, ldst);
#else
            *(bf16x8*)ldst = *(const bf16x8*)gsrc;
#endif
        }
    };

    f32x8 acc[2][4] = {};
    const bf16_t* __restrict__ arow0 = A + (size_t)(m0 + r) * lda;
    const bf16_t* __restrict__ arow1 = A + (size_t)(m0 + 16 + r) * lda;

    const int nstage = (K + KSTAGE - 1) / KSTAGE;
    stage(0, 0, (K < KSTAGE) ? K : KSTAGE);            // prologue

    for (int s = 0; s < nstage; ++s) {
        const int k0 = s * KSTAGE;
        const int kc = (K - k0 < KSTAGE) ? (K - k0) : KSTAGE;

        __syncthreads();   // everyone finished reading the buffer we overwrite
        if (s + 1 < nstage) {
            stage((s + 1) & 1, k0 + KSTAGE, KSTAGE);   // in-loop stages are full
#if USE_ASYNC_STAGE
            WAIT_ASYNC(4);  // 4 per-wave issues pending for s+1 => stage s done
#endif
        } else {
#if USE_ASYNC_STAGE
            WAIT_ASYNC(0);
#endif
        }
        __syncthreads();   // stage s visible to all waves

        const bf16_t* bbuf = bstage[s & 1];
        for (int kk = 0; kk < kc; kk += 32) {
            // A fragments (ISA 16-bit 16x32 layout; half selects K-subrange)
            union { bf16x16 v; bf16x8 h[2]; } af0, af1;
            af0.h[0] = *(const bf16x8*)(arow0 + k0 + kk + half * 8);
            af0.h[1] = *(const bf16x8*)(arow0 + k0 + kk + 16 + half * 8);
            af1.h[0] = *(const bf16x8*)(arow1 + k0 + kk + half * 8);
            af1.h[1] = *(const bf16x8*)(arow1 + k0 + kk + 16 + half * 8);
#pragma unroll
            for (int t = 0; t < 4; ++t) {
                // B fragment: lane r = column t*16+r, K split as A (half*8, 16+half*8)
                union { bf16x16 v; bf16x8 h[2]; } bu;
                bu.h[0] = *(const bf16x8*)(bbuf + (t * 16 + r) * BSTRIDE + kk + half * 8);
                bu.h[1] = *(const bf16x8*)(bbuf + (t * 16 + r) * BSTRIDE + kk + 16 + half * 8);
                bf16x16 bf = bu.v;
                acc[0][t] = __builtin_amdgcn_wmma_f32_16x16x32_bf16(
                    false, af0.v, false, bf, (short)0, acc[0][t], false, false);
                acc[1][t] = __builtin_amdgcn_wmma_f32_16x16x32_bf16(
                    false, af1.v, false, bf, (short)0, acc[1][t], false, false);
            }
        }
    }

    // Epilogue: element i of acc[mt][t] is (row = m0 + mt*16 + half*8 + i,
    // col = nbase + t*16 + r)
#pragma unroll
    for (int mt = 0; mt < 2; ++mt) {
        const int mtile = m0 + mt * 16;
#pragma unroll
        for (int t = 0; t < 4; ++t) {
            const int col = nbase + t * 16 + r;
            const float bv = BIAS ? bias[col] : 0.0f;
            if (STORE == 2) {
                bf16_t* cp = (bf16_t*)Cout + (size_t)col * ldc + mtile + half * 8;
#pragma unroll
                for (int i = 0; i < 8; ++i) {
                    float v = acc[mt][t][i] + bv;
                    if (ACT == 1) v = fmaxf(v, 0.0f);
                    cp[i] = (bf16_t)v;               // contiguous col-major
                }
            } else if (STORE == 1) {
                bf16_t* cp = (bf16_t*)Cout;
#pragma unroll
                for (int i = 0; i < 8; ++i) {
                    float v = acc[mt][t][i] + bv;
                    if (ACT == 1) v = fmaxf(v, 0.0f);
                    cp[(size_t)(mtile + half * 8 + i) * ldc + col] = (bf16_t)v;
                }
            } else {
                float* cp = (float*)Cout;
#pragma unroll
                for (int i = 0; i < 8; ++i) {
                    float v = acc[mt][t][i] + bv;
                    if (ACT == 1) v = fmaxf(v, 0.0f);
                    cp[(size_t)(mtile + half * 8 + i) * ldc + col] = v;
                }
            }
        }
    }
}

// ---------------------------------------------------------------------------
// Host-side orchestration
// ---------------------------------------------------------------------------
extern "C" void kernel_launch(void* const* d_in, const int* in_sizes, int n_in,
                              void* d_out, int out_size, void* d_ws, size_t ws_size,
                              hipStream_t stream) {
    (void)in_sizes; (void)n_in; (void)out_size; (void)ws_size;

    // ---- inputs (setup_inputs dict order; nested params flattened in order) ----
    const float* x_in  = (const float*)d_in[0];
    const float* eemb  = (const float*)d_in[1];
    const float* temb  = (const float*)d_in[2];
    /* d_in[3] = num_nodes */
    const float* es    = (const float*)d_in[4];
    const float* We    = (const float*)d_in[5];
    const float* be    = (const float*)d_in[6];
    const float* Wt    = (const float*)d_in[7];
    const float* bt    = (const float*)d_in[8];
    const float* Wz1   = (const float*)d_in[9];
    const float* bz1   = (const float*)d_in[10];
    const float* Wh1   = (const float*)d_in[11];
    const float* bh1   = (const float*)d_in[12];
    const float* lWz1  = (const float*)d_in[13];
    const float* lbz1  = (const float*)d_in[14];
    const float* lWh1  = (const float*)d_in[15];
    const float* lbh1  = (const float*)d_in[16];
    const float* Wz2   = (const float*)d_in[17];
    const float* bz2   = (const float*)d_in[18];
    const float* Wh2   = (const float*)d_in[19];
    const float* bh2   = (const float*)d_in[20];
    const float* lWz2  = (const float*)d_in[21];
    const float* lbz2  = (const float*)d_in[22];
    const float* lWh2  = (const float*)d_in[23];
    const float* lbh2  = (const float*)d_in[24];
    const float* Wmu   = (const float*)d_in[25];
    const float* bmu   = (const float*)d_in[26];
    const float* Wlv   = (const float*)d_in[27];
    const float* blv   = (const float*)d_in[28];
    const float* Wd    = (const float*)d_in[29];
    const float* bd    = (const float*)d_in[30];
    const float* Wzd   = (const float*)d_in[31];
    const float* bzd   = (const float*)d_in[32];
    const float* Whd   = (const float*)d_in[33];
    const float* bhd   = (const float*)d_in[34];
    const float* lWzd  = (const float*)d_in[35];
    const float* lbzd  = (const float*)d_in[36];
    const float* lWhd  = (const float*)d_in[37];
    const float* lbhd  = (const float*)d_in[38];

    // ---- outputs: (recon[4096,256], mu[4096,64], logvar[4096,64], adj[4096,4096]) ----
    float* out        = (float*)d_out;
    float* out_recon  = out;
    float* out_mu     = out + (size_t)NN * DIN;
    float* out_lv     = out_mu + (size_t)NN * LATD;
    float* out_adj    = out_lv + (size_t)NN * LATD;

    // ---- workspace bump allocator (256B-aligned slots) ----
    char* wsb = (char*)d_ws;
    size_t off = 0;
    auto alloc = [&](size_t bytes) -> char* {
        char* p = wsb + off;
        off += (bytes + 255) & ~(size_t)255;
        return p;
    };
    bf16_t* anT   = (bf16_t*)alloc((size_t)NN * NN * 2);      // 33.5 MB, L2-resident
    float*  dinv  = (float*)alloc(NN * 4);
    bf16_t* hbuf  = (bf16_t*)alloc((size_t)NN * DH * 2);
    bf16_t* eembB = (bf16_t*)alloc((size_t)NN * DIN * 2);
    bf16_t* tembB = (bf16_t*)alloc((size_t)NN * DIN * 2);
    bf16_t* gz    = (bf16_t*)alloc((size_t)NN * HIDD * 2);    // col-major (B operand)
    bf16_t* gh    = (bf16_t*)alloc((size_t)NN * HIDD * 2);
    bf16_t* cz    = (bf16_t*)alloc((size_t)NN * HIDD * 2);    // row-major (A operand)
    bf16_t* ch    = (bf16_t*)alloc((size_t)NN * HIDD * 2);
    float*  zl    = (float*)alloc((size_t)NN * HIDD * 4);
    float*  hl    = (float*)alloc((size_t)NN * HIDD * 4);
    bf16_t* h2    = (bf16_t*)alloc((size_t)NN * HIDD * 2);
    bf16_t* xd    = (bf16_t*)alloc((size_t)NN * DIN * 2);
    bf16_t* zs    = (bf16_t*)alloc((size_t)NN * LATD * 2);
    bf16_t* WeC   = (bf16_t*)alloc(256 * 256 * 2);
    bf16_t* WtC   = (bf16_t*)alloc(256 * 256 * 2);
    bf16_t* Wz1C  = (bf16_t*)alloc(768 * 256 * 2);
    bf16_t* Wh1C  = (bf16_t*)alloc(768 * 256 * 2);
    bf16_t* lWz1C = (bf16_t*)alloc(256 * 256 * 2);
    bf16_t* lWh1C = (bf16_t*)alloc(256 * 256 * 2);
    bf16_t* Wz2C  = (bf16_t*)alloc(256 * 256 * 2);
    bf16_t* Wh2C  = (bf16_t*)alloc(256 * 256 * 2);
    bf16_t* lWz2C = (bf16_t*)alloc(256 * 256 * 2);
    bf16_t* lWh2C = (bf16_t*)alloc(256 * 256 * 2);
    bf16_t* WmuC  = (bf16_t*)alloc(256 * 64 * 2);
    bf16_t* WlvC  = (bf16_t*)alloc(256 * 64 * 2);
    bf16_t* WdC   = (bf16_t*)alloc(64 * 256 * 2);
    bf16_t* WzdC  = (bf16_t*)alloc(256 * 256 * 2);
    bf16_t* WhdC  = (bf16_t*)alloc(256 * 256 * 2);
    bf16_t* lWzdC = (bf16_t*)alloc(256 * 256 * 2);
    bf16_t* lWhdC = (bf16_t*)alloc(256 * 256 * 2);

    // ---- graph normalization + adj output ----
    deg_kernel<<<NN / 256, 256, 0, stream>>>(es, dinv);
    adj_kernel<<<8192, 256, 0, stream>>>(es, out_adj, NN * NN);
    {
        dim3 g(NN / 32, NN / 32);
        anormt_kernel<<<g, 256, 0, stream>>>(es, dinv, anT);
    }

    // ---- bf16 packing / weight transposes ----
    const int PB = 256;
    pack_rows_kernel<<<(NN * DIN + PB - 1) / PB, PB, 0, stream>>>(x_in, hbuf, NN, DIN, DH, 0);
    pack_rows_kernel<<<(NN * DIN + PB - 1) / PB, PB, 0, stream>>>(eemb, eembB, NN, DIN, DIN, 0);
    pack_rows_kernel<<<(NN * DIN + PB - 1) / PB, PB, 0, stream>>>(temb, tembB, NN, DIN, DIN, 0);

    auto wt = [&](const float* W, bf16_t* Wc, int K, int Nn) {
        wtrans_kernel<<<(K * Nn + PB - 1) / PB, PB, 0, stream>>>(W, Wc, K, Nn);
    };
    wt(We, WeC, 256, 256);     wt(Wt, WtC, 256, 256);
    wt(Wz1, Wz1C, 768, 256);   wt(Wh1, Wh1C, 768, 256);
    wt(lWz1, lWz1C, 256, 256); wt(lWh1, lWh1C, 256, 256);   // top 256 rows of [512,256]
    wt(Wz2, Wz2C, 256, 256);   wt(Wh2, Wh2C, 256, 256);
    wt(lWz2, lWz2C, 256, 256); wt(lWh2, lWh2C, 256, 256);
    wt(Wmu, WmuC, 256, 64);    wt(Wlv, WlvC, 256, 64);
    wt(Wd, WdC, 64, 256);
    wt(Wzd, WzdC, 256, 256);   wt(Whd, WhdC, 256, 256);
    wt(lWzd, lWzdC, 256, 256); wt(lWhd, lWhdC, 256, 256);

    // ---- GEMM launch shorthands (block tile 256M x 64N) ----
    dim3 blk(256);
    auto grid = [](int M, int Nn) { return dim3(Nn / 64, M / 256); };

    // e = relu(eemb @ We + be) -> hbuf cols 256..511 ; t -> cols 512..767
    gemm_wmma_bf16<1, 1, true><<<grid(NN, 256), blk, 0, stream>>>(
        eembB, 256, WeC, 256, be, hbuf + 256, DH, NN, 256, 256);
    gemm_wmma_bf16<1, 1, true><<<grid(NN, 256), blk, 0, stream>>>(
        tembB, 256, WtC, 256, bt, hbuf + 512, DH, NN, 256, 256);

    // ---- TGCN layer: hidden = relu((1-sig(zl))*tanh(hl)) ----
    auto tgcn = [&](const bf16_t* Ain, int lda, int K,
                    const bf16_t* WzC, const float* bz,
                    const bf16_t* WhC, const float* bh,
                    const bf16_t* lWzC, const float* lbz,
                    const bf16_t* lWhC, const float* lbh,
                    void* hidden_out, bool out_bf16) {
        gemm_wmma_bf16<0, 2, false><<<grid(NN, HIDD), blk, 0, stream>>>(
            Ain, lda, WzC, K, nullptr, gz, NN, NN, HIDD, K);
        gemm_wmma_bf16<0, 2, false><<<grid(NN, HIDD), blk, 0, stream>>>(
            Ain, lda, WhC, K, nullptr, gh, NN, NN, HIDD, K);
        // dominant aggregation GEMMs (K = 4096, anormT from L2, B via async LDS)
        gemm_wmma_bf16<0, 1, true><<<grid(NN, HIDD), blk, 0, stream>>>(
            anT, NN, gz, NN, bz, cz, HIDD, NN, HIDD, NN);
        gemm_wmma_bf16<0, 1, true><<<grid(NN, HIDD), blk, 0, stream>>>(
            anT, NN, gh, NN, bh, ch, HIDD, NN, HIDD, NN);
        gemm_wmma_bf16<0, 0, true><<<grid(NN, HIDD), blk, 0, stream>>>(
            cz, HIDD, lWzC, HIDD, lbz, zl, HIDD, NN, HIDD, HIDD);
        gemm_wmma_bf16<0, 0, true><<<grid(NN, HIDD), blk, 0, stream>>>(
            ch, HIDD, lWhC, HIDD, lbh, hl, HIDD, NN, HIDD, HIDD);
        if (out_bf16)
            gate_kernel<true><<<(NN * HIDD + 255) / 256, 256, 0, stream>>>(zl, hl, hidden_out, NN * HIDD);
        else
            gate_kernel<false><<<(NN * HIDD + 255) / 256, 256, 0, stream>>>(zl, hl, hidden_out, NN * HIDD);
    };

    // tg1: input = [x|e|t] [4096,768]
    tgcn(hbuf, DH, DH, Wz1C, bz1, Wh1C, bh1, lWz1C, lbz1, lWh1C, lbh1, h2, true);
    // tg2: input = h2 [4096,256]
    tgcn(h2, HIDD, HIDD, Wz2C, bz2, Wh2C, bh2, lWz2C, lbz2, lWh2C, lbh2, h2, true);

    // mu / logvar (fp32, N = 64)
    gemm_wmma_bf16<0, 0, true><<<grid(NN, LATD), blk, 0, stream>>>(
        h2, HIDD, WmuC, HIDD, bmu, out_mu, LATD, NN, LATD, HIDD);
    gemm_wmma_bf16<0, 0, true><<<grid(NN, LATD), blk, 0, stream>>>(
        h2, HIDD, WlvC, HIDD, blv, out_lv, LATD, NN, LATD, HIDD);

    // reparameterize -> z (bf16 row-major)
    sample_kernel<<<(NN * LATD + 255) / 256, 256, 0, stream>>>(out_mu, out_lv, zs, NN * LATD);

    // decoder: xd = z @ Wd + bd (bf16 row-major, K = 64 single stage)
    gemm_wmma_bf16<0, 1, true><<<grid(NN, DIN), blk, 0, stream>>>(
        zs, LATD, WdC, LATD, bd, xd, DIN, NN, DIN, LATD);

    // tgd: input = xd -> recon (fp32 to d_out)
    tgcn(xd, DIN, DIN, WzdC, bzd, WhdC, bhd, lWzdC, lbzd, lWhdC, lbhd, out_recon, false);
}
